// GCNEncoder_86981677679215
// MI455X (gfx1250) — matrix operand order repaired and checked by reading, not compiled, not run on hardware
//
#include <hip/hip_runtime.h>
#include <hip/hip_bf16.h>

#define NN 100000
#define IN_CH 256
#define HID_CH 128
#define OUT_CH 64

static_assert(NN % 16 == 0, "M tiles must be exact");

typedef __attribute__((ext_vector_type(16))) __bf16 v16bf;
typedef __attribute__((ext_vector_type(8)))  float  v8f;
typedef __attribute__((ext_vector_type(8)))  unsigned short u16x8;

struct V16Pair { u16x8 lo, hi; };   // 32 bytes == v16bf

// round-to-nearest-even f32 -> bf16
__device__ __forceinline__ unsigned short f2bf(float f) {
  unsigned int u = __builtin_bit_cast(unsigned int, f);
  u += 0x7FFFu + ((u >> 16) & 1u);
  return (unsigned short)(u >> 16);
}

__global__ void fill_f32(float* __restrict__ p, float v, int n) {
  int i = blockIdx.x * blockDim.x + threadIdx.x;
  if (i < n) p[i] = v;
}

// deg[dst] += 1 per real edge (deg pre-filled with 1.0 for the self-loop)
__global__ void deg_accum(const int* __restrict__ dst, float* __restrict__ deg, int E) {
  int e = blockIdx.x * blockDim.x + threadIdx.x;
  if (e < E) unsafeAtomicAdd(&deg[dst[e]], 1.0f);
}

__global__ void deg_to_dinv(float* __restrict__ deg, int n) {
  int i = blockIdx.x * blockDim.x + threadIdx.x;
  if (i < n) {
    float d = deg[i];
    deg[i] = (d > 0.0f) ? rsqrtf(d) : 0.0f;
  }
}

// Pre-pack W[K,N] (f32 row-major) into bf16 B-fragment order:
// P[((kt*(N/16) + nT)*32 + lane)*16 + i] = bf16(W[(kt*32 + (lane>>4)*16 + i)*N + nT*16 + (lane&15)])
// so each lane's 16-element fragment is 32 contiguous bytes.
template <int K, int N>
__global__ void pack_w(const float* __restrict__ W, unsigned short* __restrict__ P) {
  int o = blockIdx.x * blockDim.x + threadIdx.x;
  if (o >= K * N) return;
  const int i    = o & 15;
  const int lane = (o >> 4) & 31;
  const int rest = o >> 9;
  const int nT   = rest % (N / 16);
  const int kt   = rest / (N / 16);
  const int kk   = kt * 32 + (lane >> 4) * 16 + i;
  const int n    = nT * 16 + (lane & 15);
  P[o] = f2bf(W[kk * N + n]);
}

// H[M,N] = X[M,K] @ W[K,N] using v_wmma_f32_16x16x32_bf16.
// Block = one 16-row M tile, wave w owns N-tile w; A staged in LDS as bf16,
// B fetched pre-packed (32 contiguous bytes per lane per fragment).
template <int K, int N>
__global__ __launch_bounds__(32 * (N / 16)) void gemm_wmma_bf16(
    const float* __restrict__ X, const unsigned short* __restrict__ PW,
    float* __restrict__ H) {
  __shared__ __align__(16) unsigned short ldsA[16 * K];

  const int mBase = blockIdx.x * 16;
  for (int idx = threadIdx.x; idx < 16 * K; idx += blockDim.x) {
    const int r = idx / K, c = idx % K;
    ldsA[idx] = f2bf(X[(mBase + r) * K + c]);
  }
  __syncthreads();

  const int lane = threadIdx.x & 31;
  const int wave = threadIdx.x >> 5;
  const int half = lane >> 4;
  const int l16  = lane & 15;
  const int nBase = wave * 16;

  v8f acc = {};
#pragma unroll
  for (int kb = 0; kb < K; kb += 32) {
    // A 16x32 bf16 fragment: one contiguous 16-element run per lane (k = 16*half + i)
    const unsigned short* pa = &ldsA[l16 * K + kb + half * 16];
    V16Pair av;
    av.lo = *(const u16x8*)(pa);
    av.hi = *(const u16x8*)(pa + 8);
    v16bf a = __builtin_bit_cast(v16bf, av);

    // B fragment: 32 contiguous bytes from the packed weight buffer
    const unsigned short* pb = PW + (((kb >> 5) * (N / 16) + wave) * 32 + lane) * 16;
    V16Pair bv;
    bv.lo = *(const u16x8*)(pb);
    bv.hi = *(const u16x8*)(pb + 8);
    v16bf b = __builtin_bit_cast(v16bf, bv);

    acc = __builtin_amdgcn_wmma_f32_16x16x32_bf16(false, a, false, b,
                                                  (short)0, acc, false, false);
  }

#pragma unroll
  for (int j = 0; j < 8; ++j) {
    // C/D: vgpr j -> row j (lanes 0-15) / row j+8 (lanes 16-31); lane%16 = col
    H[(mBase + half * 8 + j) * N + nBase + l16] = acc[j];
  }
}

// agg[dst] += h[src] * dinv[src]*dinv[dst]; one thread per (edge, 4-feature chunk)
template <int F>
__global__ void aggregate_edges(const int* __restrict__ src, const int* __restrict__ dst,
                                const float* __restrict__ dinv,
                                const float* __restrict__ h,
                                float* __restrict__ agg, int E) {
  const int chunks = F / 4;
  const int tid = blockIdx.x * blockDim.x + threadIdx.x;
  if (tid >= E * chunks) return;
  const int e  = tid / chunks;
  const int fc = (tid % chunks) * 4;
  const int s = src[e], d = dst[e];
  const float w = dinv[s] * dinv[d];
  const float4 v = *(const float4*)(h + s * F + fc);
  float* o = agg + d * F + fc;
  unsafeAtomicAdd(o + 0, v.x * w);
  unsafeAtomicAdd(o + 1, v.y * w);
  unsafeAtomicAdd(o + 2, v.z * w);
  unsafeAtomicAdd(o + 3, v.w * w);
}

// out = agg + h*dinv^2 (self-loop term) + bias, optional ReLU
template <int F, bool RELU>
__global__ void epilogue(const float* __restrict__ agg, const float* __restrict__ h,
                         const float* __restrict__ dinv, const float* __restrict__ bias,
                         float* __restrict__ out, int n) {
  const int tid = blockIdx.x * blockDim.x + threadIdx.x;
  if (tid >= n * F) return;
  const int i = tid / F;
  const int f = tid % F;
  const float di = dinv[i];
  float v = agg[tid] + h[tid] * (di * di) + bias[f];
  if (RELU) v = fmaxf(v, 0.0f);
  out[tid] = v;
}

static inline int ceilDiv(long long a, long long b) { return (int)((a + b - 1) / b); }

extern "C" void kernel_launch(void* const* d_in, const int* in_sizes, int n_in,
                              void* d_out, int out_size, void* d_ws, size_t ws_size,
                              hipStream_t stream) {
  const float* x  = (const float*)d_in[0];
  const int*   ei = (const int*)d_in[1];   // [2, E] row-major
  const float* W1 = (const float*)d_in[2];
  const float* b1 = (const float*)d_in[3];
  const float* W2 = (const float*)d_in[4];
  const float* b2 = (const float*)d_in[5];
  float* out = (float*)d_out;

  const int M = NN;
  const int E = in_sizes[1] / 2;
  const int* src = ei;
  const int* dst = ei + E;

  float* dinv = (float*)d_ws;                          // [N] deg -> dinv in place
  float* h1   = dinv + M;                              // [N,128]
  float* a1   = h1 + (size_t)M * HID_CH;               // [N,128]
  float* a2   = a1 + (size_t)M * HID_CH;               // [N,64]
  float* h2   = h1;                                    // reuse after layer-1 epilogue
  unsigned short* pw1 = (unsigned short*)(a2 + (size_t)M * OUT_CH);  // [256*128] bf16 packed
  unsigned short* pw2 = pw1 + IN_CH * HID_CH;                        // [128*64]  bf16 packed

  const int T = 256;

  // graph prep: deg (self-loop seeds 1.0) -> dinv; zero accumulators; pack weights
  fill_f32<<<ceilDiv(M, T), T, 0, stream>>>(dinv, 1.0f, M);
  fill_f32<<<ceilDiv((long long)M * HID_CH, T), T, 0, stream>>>(a1, 0.0f, M * HID_CH);
  fill_f32<<<ceilDiv((long long)M * OUT_CH, T), T, 0, stream>>>(a2, 0.0f, M * OUT_CH);
  pack_w<IN_CH, HID_CH><<<ceilDiv(IN_CH * HID_CH, T), T, 0, stream>>>(W1, pw1);
  pack_w<HID_CH, OUT_CH><<<ceilDiv(HID_CH * OUT_CH, T), T, 0, stream>>>(W2, pw2);
  deg_accum<<<ceilDiv(E, T), T, 0, stream>>>(dst, dinv, E);
  deg_to_dinv<<<ceilDiv(M, T), T, 0, stream>>>(dinv, M);

  // layer 1: h1 = x @ W1 ; a1 += norm-weighted scatter ; a1 = relu(a1 + dinv^2*h1 + b1)
  gemm_wmma_bf16<IN_CH, HID_CH><<<M / 16, 32 * (HID_CH / 16), 0, stream>>>(x, pw1, h1);
  aggregate_edges<HID_CH><<<ceilDiv((long long)E * (HID_CH / 4), T), T, 0, stream>>>(
      src, dst, dinv, h1, a1, E);
  epilogue<HID_CH, true><<<ceilDiv((long long)M * HID_CH, T), T, 0, stream>>>(
      a1, h1, dinv, b1, a1, M);

  // layer 2: h2 = a1 @ W2 ; a2 += norm-weighted scatter ; out = a2 + dinv^2*h2 + b2
  gemm_wmma_bf16<HID_CH, OUT_CH><<<M / 16, 32 * (OUT_CH / 16), 0, stream>>>(a1, pw2, h2);
  aggregate_edges<OUT_CH><<<ceilDiv((long long)E * (OUT_CH / 4), T), T, 0, stream>>>(
      src, dst, dinv, h2, a2, E);
  epilogue<OUT_CH, false><<<ceilDiv((long long)M * OUT_CH, T), T, 0, stream>>>(
      a2, h2, dinv, b2, out, M);
}
